// GCN_79035988181206
// MI455X (gfx1250) — matrix operand order repaired and checked by reading, not compiled, not run on hardware
//
#include <hip/hip_runtime.h>
#include <hip/hip_bf16.h>

typedef __attribute__((ext_vector_type(16))) __bf16 bf16x16;
typedef __attribute__((ext_vector_type(8)))  float  f32x8;
typedef __attribute__((ext_vector_type(4)))  float  f32x4;

#define CCH 128  // hidden channels

// ---------------------------------------------------------------- small utils

__global__ __launch_bounds__(256) void fill0_kernel(float* __restrict__ p, int n) {
  int i = blockIdx.x * 256 + threadIdx.x;
  if (i < n) p[i] = 0.f;
}

// deg[c[e]] += w[e]   (segment_sum of edge weights at destination)
__global__ __launch_bounds__(256) void deg_kernel(const int* __restrict__ col,
                                                  const float* __restrict__ w,
                                                  float* __restrict__ deg, int nE) {
  int e = blockIdx.x * 256 + threadIdx.x;
  if (e < nE) atomicAdd(&deg[col[e]], w[e]);
}

// deg -> dinv in place. Self-loop weight 1 added here, so deg >= 1 > 0 always.
__global__ __launch_bounds__(256) void dinv_kernel(float* __restrict__ deg, int n) {
  int i = blockIdx.x * 256 + threadIdx.x;
  if (i < n) deg[i] = rsqrtf(deg[i] + 1.0f);
}

// norm[e] = dinv[r[e]] * w[e] * dinv[c[e]]
__global__ __launch_bounds__(256) void norm_kernel(const int* __restrict__ row,
                                                   const int* __restrict__ col,
                                                   const float* __restrict__ w,
                                                   const float* __restrict__ dinv,
                                                   float* __restrict__ norm, int nE) {
  int e = blockIdx.x * 256 + threadIdx.x;
  if (e < nE) norm[e] = dinv[row[e]] * w[e] * dinv[col[e]];
}

// -------------------------------------------------------- WMMA GEMM  Y = op(X) @ W (+bias)(+relu)
// X: [nRows x 128] f32 (optionally gathered via idx), W: [128 x 128] f32 row-major (k-major)
// One block = 8 waves; wave handles a 16-row strip; W staged to LDS as bf16 transposed.
template<bool RELU_IN, bool RELU_OUT, bool GATHER>
__global__ __launch_bounds__(256) void gemm128_kernel(const float* __restrict__ X,
                                                      const int*   __restrict__ idx,
                                                      const float* __restrict__ W,
                                                      const float* __restrict__ bias,
                                                      float* __restrict__ Y,
                                                      int nRows) {
  __shared__ __bf16 sWT[CCH * CCH];  // sWT[n*128+k] = (bf16) W[k*128+n]   (32 KB)

  int tid = threadIdx.x;
  for (int t = tid; t < CCH * CCH; t += 256) {
    int k = t >> 7, n = t & 127;
    sWT[n * CCH + k] = (__bf16)W[t];            // coalesced global read
  }
  __syncthreads();

  int wave = tid >> 5;
  int lane = tid & 31;
  int half = lane >> 4;   // which 16-lane half
  int m    = lane & 15;

  int nStrips = (nRows + 15) >> 4;
  int strip = blockIdx.x * 8 + wave;            // wave-uniform -> EXEC stays full
  if (strip >= nStrips) return;

  // A-matrix row for this lane (ISA 16-bit A 16x32 layout: row = lane%16)
  int row  = strip * 16 + m;
  int rowL = row < nRows ? row : nRows - 1;     // clamp loads; stores guarded below
  const float* xrow = X + (GATHER ? (long)idx[rowL] * CCH : (long)rowL * CCH);

  f32x8 acc[8];
#pragma unroll
  for (int n = 0; n < 8; ++n)
#pragma unroll
    for (int j = 0; j < 8; ++j) acc[n][j] = 0.f;

#pragma unroll
  for (int kc = 0; kc < 4; ++kc) {              // K = 128 in chunks of 32
    // A layout: lanes 0-15 get K = 0..15, lanes 16-31 get K = 16..31 (matches B)
    int k0 = kc * 32 + half * 16;
    int k1 = k0 + 8;
    f32x4 f0 = *(const f32x4*)(xrow + k0);
    f32x4 f1 = *(const f32x4*)(xrow + k0 + 4);
    f32x4 f2 = *(const f32x4*)(xrow + k1);
    f32x4 f3 = *(const f32x4*)(xrow + k1 + 4);
    if (RELU_IN) {
#pragma unroll
      for (int j = 0; j < 4; ++j) {
        f0[j] = fmaxf(f0[j], 0.f); f1[j] = fmaxf(f1[j], 0.f);
        f2[j] = fmaxf(f2[j], 0.f); f3[j] = fmaxf(f3[j], 0.f);
      }
    }
    bf16x16 a;
#pragma unroll
    for (int j = 0; j < 4; ++j) {
      a[j]      = (__bf16)f0[j];
      a[4 + j]  = (__bf16)f1[j];
      a[8 + j]  = (__bf16)f2[j];
      a[12 + j] = (__bf16)f3[j];
    }
#pragma unroll
    for (int n = 0; n < 8; ++n) {               // 8 N-tiles of 16 columns
      // B layout 32x16: lane col = lane%16, lanes 0-15 K=0..15, 16-31 K=16..31 -> contiguous 16 bf16
      const __bf16* bp = &sWT[(n * 16 + m) * CCH + kc * 32 + half * 16];
      bf16x16 b = *(const bf16x16*)bp;          // two ds_load_b128
      acc[n] = __builtin_amdgcn_wmma_f32_16x16x32_bf16(
          false, a, false, b, (short)0, acc[n], false, false);
    }
  }

  // D layout: lane holds col = n*16 + m, rows M = v + 8*half
#pragma unroll
  for (int n = 0; n < 8; ++n) {
    int col = n * 16 + m;
    float bv = bias ? bias[col] : 0.f;
#pragma unroll
    for (int v = 0; v < 8; ++v) {
      int orow = strip * 16 + half * 8 + v;
      if (orow < nRows) {
        float o = acc[n][v] + bv;
        if (RELU_OUT) o = fmaxf(o, 0.f);
        Y[(long)orow * CCH + col] = o;
      }
    }
  }
}

// ------------------------------------------- self-loop + bias init: out = b + dinv[n]^2 * xw[n]
__global__ __launch_bounds__(256) void selfloop_bias_kernel(const float* __restrict__ xw,
                                                            const float* __restrict__ dinv,
                                                            const float* __restrict__ bias,
                                                            float* __restrict__ out, int nRows) {
  int t = blockIdx.x * 256 + threadIdx.x;
  int n = t >> 5, q = t & 31;                   // 32 lanes x float4 = 128 ch
  if (n >= nRows) return;
  float di = dinv[n];
  float d2 = di * di;
  f32x4 v = *(const f32x4*)(xw + (long)n * CCH + q * 4);
  f32x4 b = *(const f32x4*)(bias + q * 4);
  f32x4 o;
#pragma unroll
  for (int j = 0; j < 4; ++j) o[j] = b[j] + d2 * v[j];
  *(f32x4*)(out + (long)n * CCH + q * 4) = o;
}

// ------------------------------------------- edge scatter: out[c[e]] += norm[e] * xw[r[e]]
// One wave per edge; lane covers 4 channels. Edge metadata via readfirstlane -> scalar loads.
__global__ __launch_bounds__(256) void scatter_kernel(const int* __restrict__ row,
                                                      const int* __restrict__ col,
                                                      const float* __restrict__ norm,
                                                      const float* __restrict__ xw,
                                                      float* __restrict__ out, int nE) {
  int wave = (int)((blockIdx.x * 256 + threadIdx.x) >> 5);
  int lane = threadIdx.x & 31;
  if (wave >= nE) return;
  int e   = __builtin_amdgcn_readfirstlane(wave);
  int src = row[e];
  int dst = col[e];
  float nrm = norm[e];
  f32x4 v = *(const f32x4*)(xw + (long)src * CCH + lane * 4);   // coalesced 512B gather (L2-resident)
  float* p = out + (long)dst * CCH + lane * 4;
  atomicAdd(p + 0, nrm * v[0]);
  atomicAdd(p + 1, nrm * v[1]);
  atomicAdd(p + 2, nrm * v[2]);
  atomicAdd(p + 3, nrm * v[3]);
}

// ------------------------------------------- lin2: out[n,0:2] = x[n] @ W(128x2) + b
__global__ __launch_bounds__(256) void lin2_kernel(const float* __restrict__ X,
                                                   const float* __restrict__ W,
                                                   const float* __restrict__ b,
                                                   float* __restrict__ out, int nRows) {
  __shared__ float sW[256];
  if (threadIdx.x < 256) sW[threadIdx.x] = W[threadIdx.x];
  __syncthreads();
  int node = blockIdx.x * 8 + (threadIdx.x >> 5);
  int lane = threadIdx.x & 31;
  if (node >= nRows) return;
  f32x4 v = *(const f32x4*)(X + (long)node * CCH + lane * 4);
  float a0 = 0.f, a1 = 0.f;
#pragma unroll
  for (int j = 0; j < 4; ++j) {
    int k = lane * 4 + j;
    a0 += v[j] * sW[k * 2 + 0];
    a1 += v[j] * sW[k * 2 + 1];
  }
#pragma unroll
  for (int off = 16; off >= 1; off >>= 1) {
    a0 += __shfl_xor(a0, off);
    a1 += __shfl_xor(a1, off);
  }
  if (lane == 0) {
    out[(long)node * 2 + 0] = a0 + b[0];
    out[(long)node * 2 + 1] = a1 + b[1];
  }
}

// ---------------------------------------------------------------- launcher

extern "C" void kernel_launch(void* const* d_in, const int* in_sizes, int n_in,
                              void* d_out, int out_size, void* d_ws, size_t ws_size,
                              hipStream_t stream) {
  const int*   x_idx   = (const int*)d_in[0];
  const int*   e_index = (const int*)d_in[1];
  const float* e_w     = (const float*)d_in[2];
  const float* emb     = (const float*)d_in[3];
  const float* conv_w  = (const float*)d_in[4];
  const float* conv_b  = (const float*)d_in[5];
  const float* lin1_w  = (const float*)d_in[6];
  const float* lin1_b  = (const float*)d_in[7];
  const float* lin2_w  = (const float*)d_in[8];
  const float* lin2_b  = (const float*)d_in[9];
  float* out = (float*)d_out;

  const int N = in_sizes[0];
  const int E = in_sizes[2];
  const int* row = e_index;
  const int* col = e_index + E;

  // workspace layout
  auto align = [](size_t x) { return (x + 511) & ~(size_t)511; };
  char* ws = (char*)d_ws;
  size_t o = 0;
  float* dinv = (float*)(ws + o); o += align((size_t)N * 4);
  float* norm = (float*)(ws + o); o += align((size_t)E * 4);
  float* bufA = (float*)(ws + o); o += align((size_t)N * CCH * 4);
  float* bufB = (float*)(ws + o); o += align((size_t)N * CCH * 4);
  float* bufC = (float*)(ws + o); o += align((size_t)N * CCH * 4);
  (void)ws_size; (void)n_in; (void)out_size;

  const int gN    = (N + 255) / 256;
  const int gE    = (E + 255) / 256;
  const int gRow  = (N * 32 + 255) / 256;     // 32 threads per node
  const int gGemm = (((N + 15) / 16) + 7) / 8; // 8 strips per block
  const int gEdge = (E + 7) / 8;               // 1 wave per edge
  const int gNode = (N + 7) / 8;               // 1 wave per node

  // normalization
  fill0_kernel<<<gN, 256, 0, stream>>>(dinv, N);
  deg_kernel<<<gE, 256, 0, stream>>>(col, e_w, dinv, E);
  dinv_kernel<<<gN, 256, 0, stream>>>(dinv, N);
  norm_kernel<<<gE, 256, 0, stream>>>(row, col, e_w, dinv, norm, E);

  // layer 0: x = emb[x_idx];  xw = x @ W0  (gather fused into GEMM)
  gemm128_kernel<false, false, true><<<gGemm, 256, 0, stream>>>(
      emb, x_idx, conv_w + 0 * CCH * CCH, nullptr, bufB, N);
  selfloop_bias_kernel<<<gRow, 256, 0, stream>>>(bufB, dinv, conv_b + 0 * CCH, bufC, N);
  scatter_kernel<<<gEdge, 256, 0, stream>>>(row, col, norm, bufB, bufC, E);

  // layer 1: relu fused into GEMM input
  gemm128_kernel<true, false, false><<<gGemm, 256, 0, stream>>>(
      bufC, nullptr, conv_w + 1 * CCH * CCH, nullptr, bufA, N);
  selfloop_bias_kernel<<<gRow, 256, 0, stream>>>(bufA, dinv, conv_b + 1 * CCH, bufB, N);
  scatter_kernel<<<gEdge, 256, 0, stream>>>(row, col, norm, bufA, bufB, E);

  // layer 2
  gemm128_kernel<true, false, false><<<gGemm, 256, 0, stream>>>(
      bufB, nullptr, conv_w + 2 * CCH * CCH, nullptr, bufC, N);
  selfloop_bias_kernel<<<gRow, 256, 0, stream>>>(bufC, dinv, conv_b + 2 * CCH, bufA, N);
  scatter_kernel<<<gEdge, 256, 0, stream>>>(row, col, norm, bufC, bufA, E);

  // lin1: relu(x @ W + b) with bias+relu fused into GEMM epilogue (no relu on input)
  gemm128_kernel<false, true, false><<<gGemm, 256, 0, stream>>>(
      bufA, nullptr, lin1_w, lin1_b, bufB, N);

  // lin2: [N,128] @ [128,2] + b
  lin2_kernel<<<gNode, 256, 0, stream>>>(bufB, lin2_w, lin2_b, out, N);
}
